// VectorAttention_43353399885958
// MI455X (gfx1250) — compile-verified, matrix operand order checked by reading, not run on hardware
//
#include <hip/hip_runtime.h>
#include <hip/hip_bf16.h>
#include <math.h>

typedef _Float16 h16;
typedef __attribute__((ext_vector_type(16))) _Float16     v16h;
typedef __attribute__((ext_vector_type(8)))  float        v8f;
typedef __attribute__((ext_vector_type(4)))  float        v4f;
typedef __attribute__((ext_vector_type(4)))  unsigned int v4u;

union FragU { v16h v; v4u q[2]; };   // 32 bytes; trivial types -> valid union

#define B_ 8
#define N_ 384
#define F_ 64
#define H_ 128
#define LN_EPS 1e-5f
#define NCHUNK (N_ / 16)   // 24 chunks of 16 j's per (b,i)

// K striping of the 16-bit A/B fragment layout: element e of the v16h held by a lane
// maps to local K = ((e<8)? e : e+8) + 8*(lane>=16); row/col index = lane&15.
__device__ __forceinline__ int frag_k(int e, int half) {
  return ((e < 8) ? e : e + 8) + (half << 3);
}

// ---------------- LDS-resident value-net weights + closed-form LN stats ----------------
struct ValueLds {
  v4f   P[H_];                       // {w0*g, w1*g, b1*g, g} per k (ds_load_b128)
  float bt[H_];                      // LN beta
  float vb2[F_];
  float stats[9];                    // S0,S1,Sb, A00,A11,A01, B0,B1,Cb
  alignas(16) h16 vW2frag[16][32][16];  // B-frags, fragment-major: frag = t*4+nt
};

__device__ __forceinline__ void preload_value(
    const float* __restrict__ vW1, const float* __restrict__ vb1,
    const float* __restrict__ vg,  const float* __restrict__ vbeta,
    const float* __restrict__ vW2, const float* __restrict__ vb2,
    ValueLds& L, int tid, int nthr) {
  for (int k = tid; k < H_; k += nthr) {
    const float w0 = vW1[k], w1 = vW1[H_ + k], b1v = vb1[k], gv = vg[k];
    v4f p; p.x = w0 * gv; p.y = w1 * gv; p.z = b1v * gv; p.w = gv;
    L.P[k] = p;
    L.bt[k] = vbeta[k];
  }
  for (int f = tid; f < F_; f += nthr) L.vb2[f] = vb2[f];
  // pre-pack vW2 (K=H x N=F) into WMMA B fragments: 2x ds_store_b128 per lane-row
  for (int idx = tid; idx < 16 * 32; idx += nthr) {
    const int frag = idx >> 5, ln = idx & 31;
    const int t = frag >> 2, nt = frag & 3;
    const int lhalf = ln >> 4, llo = ln & 15;
    FragU fr;
#pragma unroll
    for (int e = 0; e < 16; ++e) {
      const int k = 32 * t + frag_k(e, lhalf);
      fr.v[e] = (h16)vW2[k * F_ + nt * 16 + llo];
    }
    v4u* dst = (v4u*)&L.vW2frag[frag][ln][0];
    dst[0] = fr.q[0]; dst[1] = fr.q[1];
  }
  if (tid < 9) L.stats[tid] = 0.0f;
  __syncthreads();
  if (tid < H_) {            // weight-only sums -> closed-form LayerNorm statistics
    const float a = vW1[tid], c = vW1[H_ + tid], d = vb1[tid];
    atomicAdd(&L.stats[0], a);     atomicAdd(&L.stats[1], c);
    atomicAdd(&L.stats[2], d);     atomicAdd(&L.stats[3], a * a);
    atomicAdd(&L.stats[4], c * c); atomicAdd(&L.stats[5], a * c);
    atomicAdd(&L.stats[6], a * d); atomicAdd(&L.stats[7], c * d);
    atomicAdd(&L.stats[8], d * d);
  }
  __syncthreads();
}

// Compute joined(16x64) for pairs (i, j0..j0+15): invariants -> value_net (WMMA GEMM1)
// -> merge/join. Result left in C-layout acc[4] (v8f each) and staged to LDS as f16.
__device__ __forceinline__ void compute_joined_tile(
    const float* __restrict__ inputs, const float* __restrict__ positions,
    const ValueLds& L, h16* __restrict__ stage /* [16*F_], 16B-aligned */,
    int b, int i, int j0, int lane, v8f acc[4]) {
  const int lo   = lane & 15;
  const int half = lane >> 4;
  const int j    = j0 + lo;

  __builtin_prefetch(&inputs[((size_t)b * N_ + j) * F_], 0, 3);  // global_prefetch_b8

  // pairwise geometric invariants for row m = lo (pair (i, j0+lo)), exact f32
  const float pix = positions[((size_t)b * N_ + i) * 3 + 0];
  const float piy = positions[((size_t)b * N_ + i) * 3 + 1];
  const float piz = positions[((size_t)b * N_ + i) * 3 + 2];
  const float pjx = positions[((size_t)b * N_ + j) * 3 + 0];
  const float pjy = positions[((size_t)b * N_ + j) * 3 + 1];
  const float pjz = positions[((size_t)b * N_ + j) * 3 + 2];
  const float dot = pix * pjx + piy * pjy + piz * pjz;
  const float cx = pjy * piz - pjz * piy;
  const float cy = pjz * pix - pjx * piz;
  const float cz = pjx * piy - pjy * pix;
  const float cn = sqrtf(cx * cx + cy * cy + cz * cz);

  // closed-form LayerNorm stats (h = dot*w0 + cn*w1 + b1 is affine in dot,cn)
  const float invH = 1.0f / (float)H_;
  const float mu = (dot * L.stats[0] + cn * L.stats[1] + L.stats[2]) * invH;
  const float e2 = (dot * dot * L.stats[3] + cn * cn * L.stats[4] +
                    2.f * dot * cn * L.stats[5] + 2.f * dot * L.stats[6] +
                    2.f * cn * L.stats[7] + L.stats[8]) * invH;
  const float rs = rsqrtf(e2 - mu * mu + LN_EPS);
  // relu(LN(h))*... = a1*P.x + a2*P.y + a3*P.z + a4*P.w + bt[k]
  const float a1 = rs * dot, a2 = rs * cn, a3 = rs, a4 = -rs * mu;

#pragma unroll
  for (int nt = 0; nt < 4; ++nt) { v8f z = {0, 0, 0, 0, 0, 0, 0, 0}; acc[nt] = z; }

  // GEMM1: relu(LN(h))(16x128) @ vW2(128x64); A-frag built in-register, B-frag 2x b128
#pragma unroll
  for (int t = 0; t < 4; ++t) {
    FragU af;
#pragma unroll
    for (int e = 0; e < 16; ++e) {
      const int k = 32 * t + frag_k(e, half);
      const v4f p = L.P[k];
      const float hn = a1 * p.x + a2 * p.y + a3 * p.z + a4 * p.w + L.bt[k];
      af.v[e] = (h16)fmaxf(hn, 0.0f);
    }
#pragma unroll
    for (int nt = 0; nt < 4; ++nt) {
      FragU bf;
      const v4u* bsrc = (const v4u*)&L.vW2frag[t * 4 + nt][lane][0];
      bf.q[0] = bsrc[0]; bf.q[1] = bsrc[1];
      acc[nt] = __builtin_amdgcn_wmma_f32_16x16x32_f16(false, af.v, false, bf.v,
                                                       (short)0, acc[nt], false, false);
    }
  }

  // joined = 0.5*(D + vb2 + 0.5*(in[b,j] + in[b,i]));  stage as f16 for GEMM2 A-frags
  const float* in_i = &inputs[((size_t)b * N_ + i) * F_];
#pragma unroll
  for (int nt = 0; nt < 4; ++nt) {
    const int f = nt * 16 + lo;
    const float xi = in_i[f];
#pragma unroll
    for (int r = 0; r < 8; ++r) {
      const int m = r + (half << 3);
      const float xj = inputs[((size_t)b * N_ + j0 + m) * F_ + f];
      const float jn = 0.5f * (acc[nt][r] + L.vb2[f] + 0.5f * (xi + xj));
      acc[nt][r] = jn;
      stage[m * F_ + f] = (h16)jn;
    }
  }
  asm volatile("s_wait_dscnt 0" ::: "memory");  // DS is in-order per wave; make explicit
}

// ---------------- pass 1: raw scores + per-batch max (monotone uint key) ----------------
__global__ void __launch_bounds__(256) va_scores_kernel(
    const float* __restrict__ inputs, const float* __restrict__ positions,
    const float* __restrict__ vW1, const float* __restrict__ vb1,
    const float* __restrict__ vg,  const float* __restrict__ vbeta,
    const float* __restrict__ vW2, const float* __restrict__ vb2,
    const float* __restrict__ sW1, const float* __restrict__ sb1,
    const float* __restrict__ sW2, const float* __restrict__ sb2,
    float* __restrict__ att_raw, unsigned* __restrict__ maxkey) {
  __shared__ ValueLds L;
  __shared__ alignas(16) h16 sW1frag[16][32][16];  // B-frags for sW1: frag = t*8+nt
  __shared__ float ssb1[H_], ssW2[H_];
  __shared__ float ssb2;
  __shared__ alignas(16) h16 stage[8][16 * F_];

  const int tid = threadIdx.x;
  const int b = blockIdx.x / N_;
  const int i = blockIdx.x - b * N_;
  const int lane = tid & 31, wave = tid >> 5;
  const int lo = lane & 15, half = lane >> 4;

  // pre-pack sW1 (K=F x N=H) into WMMA B fragments
  for (int idx = tid; idx < 16 * 32; idx += 256) {
    const int frag = idx >> 5, ln = idx & 31;
    const int t = frag >> 3, nt = frag & 7;
    const int lhalf = ln >> 4, llo = ln & 15;
    FragU fr;
#pragma unroll
    for (int e = 0; e < 16; ++e) {
      const int k = 32 * t + frag_k(e, lhalf);
      fr.v[e] = (h16)sW1[k * H_ + nt * 16 + llo];
    }
    v4u* dst = (v4u*)&sW1frag[frag][ln][0];
    dst[0] = fr.q[0]; dst[1] = fr.q[1];
  }
  for (int k = tid; k < H_; k += 256) { ssb1[k] = sb1[k]; ssW2[k] = sW2[k]; }
  if (tid == 0) ssb2 = sb2[0];
  preload_value(vW1, vb1, vg, vbeta, vW2, vb2, L, tid, 256);  // includes __syncthreads

  float wavemax = -3.0e38f;
  for (int c = wave; c < NCHUNK; c += 8) {       // uniform trip count across waves
    const int j0 = c * 16;
    v8f acc1[4];
    compute_joined_tile(inputs, positions, L, stage[wave], b, i, j0, lane, acc1);

    // GEMM2: joined(16x64) @ sW1(64x128); A/B frags are 2x ds_load_b128 each
    v8f acc2[8];
#pragma unroll
    for (int nt = 0; nt < 8; ++nt) { v8f z = {0, 0, 0, 0, 0, 0, 0, 0}; acc2[nt] = z; }
#pragma unroll
    for (int t = 0; t < 2; ++t) {
      FragU af;
      const h16* asrc = stage[wave] + lo * F_ + 32 * t + (half << 3);
      af.q[0] = *(const v4u*)asrc;          // K = 32t+8*half+{0..7}
      af.q[1] = *(const v4u*)(asrc + 16);   // K = 32t+8*half+16+{0..7}
#pragma unroll
      for (int nt = 0; nt < 8; ++nt) {
        FragU bf;
        const v4u* bsrc = (const v4u*)&sW1frag[t * 8 + nt][lane][0];
        bf.q[0] = bsrc[0]; bf.q[1] = bsrc[1];
        acc2[nt] = __builtin_amdgcn_wmma_f32_16x16x32_f16(false, af.v, false, bf.v,
                                                          (short)0, acc2[nt], false, false);
      }
    }

    // score[m] = sum_k relu(D + sb1[k]) * sW2[k] + sb2  (butterfly over 16 lanes/half)
    float sc[8];
#pragma unroll
    for (int r = 0; r < 8; ++r) {
      float s = 0.0f;
#pragma unroll
      for (int nt = 0; nt < 8; ++nt) {
        const int k = nt * 16 + lo;
        s += fmaxf(acc2[nt][r] + ssb1[k], 0.0f) * ssW2[k];
      }
#pragma unroll
      for (int m = 1; m < 16; m <<= 1) s += __shfl_xor(s, m, 32);
      sc[r] = s + ssb2;
    }

    const size_t base = ((size_t)b * N_ + i) * N_ + j0;
    if (lo == 0) {
#pragma unroll
      for (int r = 0; r < 8; ++r) att_raw[base + r + (half << 3)] = sc[r];
    }
    float lm = sc[0];
#pragma unroll
    for (int r = 1; r < 8; ++r) lm = fmaxf(lm, sc[r]);
    lm = fmaxf(lm, __shfl_xor(lm, 16, 32));
    wavemax = fmaxf(wavemax, lm);
  }
  if (lane == 0) {
    const unsigned bits = __float_as_uint(wavemax);
    const unsigned key = (bits & 0x80000000u) ? ~bits : (bits | 0x80000000u);
    atomicMax(&maxkey[b], key);
  }
}

__device__ __forceinline__ float key_to_float(unsigned key) {
  const unsigned bits = (key & 0x80000000u) ? (key & 0x7FFFFFFFu) : ~key;
  return __uint_as_float(bits);
}

// ---------------- pass 2: per-batch exp-sum ----------------
__global__ void __launch_bounds__(256) va_sumexp_kernel(
    const float* __restrict__ att_raw, const unsigned* __restrict__ maxkey,
    float* __restrict__ sums) {
  __shared__ float red[256];
  const int b = blockIdx.x;
  const float mx = key_to_float(maxkey[b]);
  const size_t base = (size_t)b * (N_ * N_);
  float acc = 0.0f;
  for (int p = threadIdx.x; p < N_ * N_; p += 256)
    acc += __expf(att_raw[base + p] - mx);
  red[threadIdx.x] = acc;
  __syncthreads();
  for (int s = 128; s > 0; s >>= 1) {
    if (threadIdx.x < s) red[threadIdx.x] += red[threadIdx.x + s];
    __syncthreads();
  }
  if (threadIdx.x == 0) sums[b] = red[0];
}

// ---------------- pass 3: in-place softmax normalize ----------------
__global__ void va_norm_kernel(float* __restrict__ att,
                               const unsigned* __restrict__ maxkey,
                               const float* __restrict__ sums) {
  const int idx = blockIdx.x * blockDim.x + threadIdx.x;
  if (idx >= B_ * N_ * N_) return;
  const int b = idx / (N_ * N_);
  att[idx] = __expf(att[idx] - key_to_float(maxkey[b])) * (1.0f / sums[b]);
}

// ---------------- pass 4: x = sum_j att * joined (recompute joined, GEMM1 only) ----------------
__global__ void __launch_bounds__(256) va_xout_kernel(
    const float* __restrict__ inputs, const float* __restrict__ positions,
    const float* __restrict__ vW1, const float* __restrict__ vb1,
    const float* __restrict__ vg,  const float* __restrict__ vbeta,
    const float* __restrict__ vW2, const float* __restrict__ vb2,
    const float* __restrict__ att, float* __restrict__ x_out) {
  __shared__ ValueLds L;
  __shared__ alignas(16) h16 stage[8][16 * F_];
  __shared__ float xacc[F_];

  const int tid = threadIdx.x;
  const int b = blockIdx.x / N_;
  const int i = blockIdx.x - b * N_;
  const int lane = tid & 31, wave = tid >> 5;
  const int lo = lane & 15, half = lane >> 4;

  if (tid < F_) xacc[tid] = 0.0f;
  preload_value(vW1, vb1, vg, vbeta, vW2, vb2, L, tid, 256);  // barriers cover xacc init

  float xp[4] = {0, 0, 0, 0};
  for (int c = wave; c < NCHUNK; c += 8) {
    const int j0 = c * 16;
    v8f acc1[4];
    compute_joined_tile(inputs, positions, L, stage[wave], b, i, j0, lane, acc1);
    const size_t abase = ((size_t)b * N_ + i) * N_ + j0;
    float aw[8];
#pragma unroll
    for (int r = 0; r < 8; ++r) aw[r] = att[abase + r + (half << 3)];
#pragma unroll
    for (int nt = 0; nt < 4; ++nt) {
      float s = 0.0f;
#pragma unroll
      for (int r = 0; r < 8; ++r) s += aw[r] * acc1[nt][r];
      xp[nt] += s;
    }
  }
#pragma unroll
  for (int nt = 0; nt < 4; ++nt) {
    xp[nt] += __shfl_xor(xp[nt], 16, 32);     // merge the two M-halves (same f)
    if (half == 0) atomicAdd(&xacc[nt * 16 + lo], xp[nt]);  // ds_add_f32 across waves
  }
  __syncthreads();
  if (tid < F_) x_out[((size_t)b * N_ + i) * F_ + tid] = xacc[tid];
}

// ---------------- init: per-batch max key / sums in workspace ----------------
__global__ void va_init_kernel(unsigned* __restrict__ maxkey, float* __restrict__ sums) {
  const int t = threadIdx.x;
  if (t < B_) { maxkey[t] = 0u; sums[t] = 0.0f; }
}

extern "C" void kernel_launch(void* const* d_in, const int* in_sizes, int n_in,
                              void* d_out, int out_size, void* d_ws, size_t ws_size,
                              hipStream_t stream) {
  (void)in_sizes; (void)n_in; (void)out_size; (void)ws_size;
  const float* inputs    = (const float*)d_in[0];
  const float* positions = (const float*)d_in[1];
  const float* vW1   = (const float*)d_in[2];
  const float* vb1   = (const float*)d_in[3];
  const float* vg    = (const float*)d_in[4];
  const float* vbeta = (const float*)d_in[5];
  const float* vW2   = (const float*)d_in[6];
  const float* vb2   = (const float*)d_in[7];
  const float* sW1   = (const float*)d_in[8];
  const float* sb1   = (const float*)d_in[9];
  const float* sW2   = (const float*)d_in[10];
  const float* sb2   = (const float*)d_in[11];

  float* x_out = (float*)d_out;                       // (B,N,F)
  float* att   = x_out + (size_t)B_ * N_ * F_;        // (B,N,N) raw scores -> softmax in place
  unsigned* maxkey = (unsigned*)d_ws;                 // [B]
  float*    sums   = (float*)d_ws + B_;               // [B]

  va_init_kernel<<<1, 32, 0, stream>>>(maxkey, sums);
  va_scores_kernel<<<dim3(B_ * N_), dim3(256), 0, stream>>>(
      inputs, positions, vW1, vb1, vg, vbeta, vW2, vb2, sW1, sb1, sW2, sb2, att, maxkey);
  va_sumexp_kernel<<<dim3(B_), dim3(256), 0, stream>>>(att, maxkey, sums);
  const int tot = B_ * N_ * N_;
  va_norm_kernel<<<dim3((tot + 255) / 256), dim3(256), 0, stream>>>(att, maxkey, sums);
  va_xout_kernel<<<dim3(B_ * N_), dim3(256), 0, stream>>>(
      inputs, positions, vW1, vb1, vg, vbeta, vW2, vb2, att, x_out);
}
